// TemporalGRAMEncoder_841813590027
// MI455X (gfx1250) — compile-verified, matrix operand order checked by reading, not run on hardware
//
#include <hip/hip_runtime.h>
#include <hip/hip_bf16.h>

// ---------------------------------------------------------------------------
// TemporalGRAMEncoder forward for MI455X (gfx1250, wave32, WMMA).
// Heavy GEMMs run on v_wmma_f32_16x16x32_f16 (f16 operands, f32 accum),
// register-blocked 1x4 tiles per wave. sched_group_barrier pins all 12 VMEM
// loads of a k-step ahead of the 4 WMMAs so loadcnt waits are progressive
// (one L2 round-trip per k-step) instead of a full drain per WMMA.
// ---------------------------------------------------------------------------

typedef __attribute__((ext_vector_type(16))) _Float16 v16h;
typedef __attribute__((ext_vector_type(8)))  _Float16 v8h;
typedef __attribute__((ext_vector_type(8)))  float    v8f;

#define GF_BIAS    1
#define GF_LOADC   2
#define GF_GELU    4
#define GF_CLAMP10 8

#if defined(__has_builtin)
#if __has_builtin(__builtin_amdgcn_sched_group_barrier)
#define SCHED_GROUP(mask, size) __builtin_amdgcn_sched_group_barrier((mask), (size), 0)
#endif
#endif
#ifndef SCHED_GROUP
#define SCHED_GROUP(mask, size)
#endif

__device__ __forceinline__ float gelu_exact(float x) {
    return 0.5f * x * (1.0f + erff(x * 0.70710678118654752f));
}
__device__ __forceinline__ float sigmoidf(float x) {
    return 1.0f / (1.0f + __expf(-x));
}

// ---------------------------------------------------------------------------
// WMMA GEMM:  C[M x N] = A[M x K](f32) * Bt[N x K](f16, pre-transposed)
// One 16x64 output slab per wave32 (1 M-tile x 4 N-tiles).
// M % 16 == 0, N % 64 == 0, K % 32 == 0 for every GEMM in this workload.
// Fragment layouts per CDNA5 ISA 7.12.2:
//   A: lane L -> row M = L&15; half = L>>4 selects K base 0/8 (and 16/24).
//   B: symmetric, lane L -> col N = L&15.
//   C/D: lane L -> col N = L&15; VGPR i -> row (L>>4)*8 + i.
// ---------------------------------------------------------------------------
template<int FLAGS>
__global__ __launch_bounds__(256)
void k_gemm_wmma(const float* __restrict__ A, int lda,
                 const _Float16* __restrict__ Bt, int K,
                 const float* __restrict__ bias,
                 float* __restrict__ C, int ldc,
                 int M, int groupsN)          // groupsN = N / 64
{
    int wid  = (blockIdx.x * blockDim.x + threadIdx.x) >> 5;
    int lane = threadIdx.x & 31;
    int tm   = wid / groupsN;
    int gn   = wid - tm * groupsN;
    if (tm * 16 >= M) return;                 // uniform per wave

    const int hi = lane >> 4;                 // 0 | 1 (half-wave K split)
    const int r  = lane & 15;

    const float* Arow = A + (size_t)(tm * 16 + r) * lda;
    const _Float16* Brow[4];
#pragma unroll
    for (int j = 0; j < 4; ++j)
        Brow[j] = Bt + (size_t)(gn * 64 + j * 16 + r) * (size_t)K;

    v8f acc[4];
#pragma unroll
    for (int j = 0; j < 4; ++j) acc[j] = (v8f){};
    if (FLAGS & GF_LOADC) {
#pragma unroll
        for (int j = 0; j < 4; ++j) {
            int col = gn * 64 + j * 16 + r;
#pragma unroll
            for (int i = 0; i < 8; ++i)
                acc[j][i] = C[(size_t)(tm * 16 + hi * 8 + i) * ldc + col];
        }
    }

    for (int k0 = 0; k0 < K; k0 += 32) {
        const int kb = k0 + hi * 8;

        // ---- issue every VMEM load of this k-step before any consumer ----
        float4 a0 = *(const float4*)(Arow + kb);
        float4 a1 = *(const float4*)(Arow + kb + 4);
        float4 a2 = *(const float4*)(Arow + kb + 16);
        float4 a3 = *(const float4*)(Arow + kb + 20);
        v8h bl[4], bh[4];
#pragma unroll
        for (int j = 0; j < 4; ++j) {
            bl[j] = *(const v8h*)(Brow[j] + kb);
            bh[j] = *(const v8h*)(Brow[j] + kb + 16);
        }

        // ---- convert A to f16 fragment (overlaps with B loads in flight) --
        v16h a;
        a[0]  = (_Float16)a0.x; a[1]  = (_Float16)a0.y;
        a[2]  = (_Float16)a0.z; a[3]  = (_Float16)a0.w;
        a[4]  = (_Float16)a1.x; a[5]  = (_Float16)a1.y;
        a[6]  = (_Float16)a1.z; a[7]  = (_Float16)a1.w;
        a[8]  = (_Float16)a2.x; a[9]  = (_Float16)a2.y;
        a[10] = (_Float16)a2.z; a[11] = (_Float16)a2.w;
        a[12] = (_Float16)a3.x; a[13] = (_Float16)a3.y;
        a[14] = (_Float16)a3.z; a[15] = (_Float16)a3.w;

        // ---- 4 back-to-back WMMAs on the shared A fragment ----------------
#pragma unroll
        for (int j = 0; j < 4; ++j) {
            v16h b;
#pragma unroll
            for (int q = 0; q < 8; ++q) { b[q] = bl[j][q]; b[8 + q] = bh[j][q]; }
            acc[j] = __builtin_amdgcn_wmma_f32_16x16x32_f16(
                         false, a, false, b, (short)0, acc[j], false, false);
        }

        // ---- scheduling pipeline for this k-step:
        //      all 12 VMEM reads first, then VALU cvts, then the 4 WMMAs ----
        SCHED_GROUP(0x020, 12);   // VMEM read group
        SCHED_GROUP(0x002, 10);   // VALU group (8 cvt_pk + addr math)
        SCHED_GROUP(0x008, 4);    // MFMA/WMMA group
    }

#pragma unroll
    for (int j = 0; j < 4; ++j) {
        const int col = gn * 64 + j * 16 + r;
        const float bv = (FLAGS & GF_BIAS) ? bias[col] : 0.0f;
#pragma unroll
        for (int i = 0; i < 8; ++i) {
            float v = acc[j][i] + bv;
            if (FLAGS & GF_GELU)    v = gelu_exact(v);
            if (FLAGS & GF_CLAMP10) v = fminf(v, 10.0f);
            C[(size_t)(tm * 16 + hi * 8 + i) * ldc + col] = v;
        }
    }
}

// ---------------------------------------------------------------------------
// Small utility / pointwise kernels
// ---------------------------------------------------------------------------
__global__ void k_fill_f32(float* p, float v, long long n) {
    long long t = (long long)blockIdx.x * blockDim.x + threadIdx.x;
    if (t < n) p[t] = v;
}
__global__ void k_copy_f32(const float* __restrict__ in, float* __restrict__ out, int n) {
    int t = blockIdx.x * blockDim.x + threadIdx.x;
    if (t < n) out[t] = in[t];
}
__global__ void k_cast_f16(const float* __restrict__ in, _Float16* __restrict__ out, long long n) {
    long long t = (long long)blockIdx.x * blockDim.x + threadIdx.x;
    if (t < n) out[t] = (_Float16)in[t];
}
// in[K][N] row-major f32  ->  out[N][K] row-major f16
__global__ void k_cast_transpose_f16(const float* __restrict__ in, _Float16* __restrict__ out,
                                     int K, int N) {
    long long t = (long long)blockIdx.x * blockDim.x + threadIdx.x;
    if (t >= (long long)K * N) return;
    int k = (int)(t / N);
    int n = (int)(t - (long long)k * N);
    out[(size_t)n * K + k] = (_Float16)in[t];
}
__global__ void k_vadd(const float* __restrict__ a, const float* __restrict__ b,
                       float* __restrict__ o, int n) {
    int t = blockIdx.x * blockDim.x + threadIdx.x;
    if (t < n) o[t] = a[t] + b[t];
}
__global__ void k_gelu_inplace(float* p, long long n) {
    long long t = (long long)blockIdx.x * blockDim.x + threadIdx.x;
    if (t < n) p[t] = gelu_exact(p[t]);
}

// ---------------------------------------------------------------------------
// Graph kernels
// ---------------------------------------------------------------------------
__global__ void k_deg_edges(const int* __restrict__ dst, float* __restrict__ deg, int E) {
    int t = blockIdx.x * blockDim.x + threadIdx.x;
    if (t < E) atomicAdd(&deg[dst[t]], 1.0f);
}
__global__ void k_rsqrt_inplace(float* p, int n) {
    int t = blockIdx.x * blockDim.x + threadIdx.x;
    if (t < n) p[t] = rsqrtf(p[t]);
}
__global__ void k_edge_norm(const int* __restrict__ src, const int* __restrict__ dst,
                            const float* __restrict__ dinv, float* __restrict__ nrm, int E) {
    int t = blockIdx.x * blockDim.x + threadIdx.x;
    if (t < E) nrm[t] = dinv[src[t]] * dinv[dst[t]];
}
// out[i][f] = bias[f] + dinv[i]^2 * xw[i][f]   (self-loop + bias init)
__global__ void k_conv_init(const float* __restrict__ xw, const float* __restrict__ dinv,
                            const float* __restrict__ bias, float* __restrict__ out,
                            long long total) {
    long long t = (long long)blockIdx.x * blockDim.x + threadIdx.x;
    if (t >= total) return;
    int i = (int)(t >> 9);
    int f = (int)(t & 511);
    float d = dinv[i];
    out[t] = bias[f] + d * d * xw[t];
}
// out[dst[e]][f..f+3] += norm[e] * xw[src[e]][f..f+3]
__global__ void k_conv_edges(const float* __restrict__ xw, const float* __restrict__ nrm,
                             const int* __restrict__ src, const int* __restrict__ dst,
                             float* __restrict__ out, long long total) {
    long long t = (long long)blockIdx.x * blockDim.x + threadIdx.x;
    if (t >= total) return;
    int e  = (int)(t >> 7);
    int f4 = (int)(t & 127) << 2;
    float  w = nrm[e];
    float4 x = *(const float4*)(xw + (size_t)src[e] * 512 + f4);
    float* o = out + (size_t)dst[e] * 512 + f4;
    atomicAdd(o + 0, w * x.x);
    atomicAdd(o + 1, w * x.y);
    atomicAdd(o + 2, w * x.z);
    atomicAdd(o + 3, w * x.w);
}
// pooled[batch[i]][f] += h[i][f]
__global__ void k_pool(const float* __restrict__ h, const int* __restrict__ batch,
                       float* __restrict__ pooled_t, long long total) {
    long long t = (long long)blockIdx.x * blockDim.x + threadIdx.x;
    if (t >= total) return;
    int i = (int)(t >> 9);
    int f = (int)(t & 511);
    atomicAdd(&pooled_t[(size_t)batch[i] * 512 + f], h[t]);
}

// ---------------------------------------------------------------------------
// LSTM gate pointwise:  G[B x 2048] (i|f|g|o), updates h,c [B x 512]
// ---------------------------------------------------------------------------
__global__ void k_lstm_gates(const float* __restrict__ G, float* __restrict__ h,
                             float* __restrict__ c, float* __restrict__ ys, int total) {
    int t = blockIdx.x * blockDim.x + threadIdx.x;
    if (t >= total) return;
    int b = t >> 9;
    int j = t & 511;
    const float* g = G + (size_t)b * 2048;
    float ig = g[j], fg = g[512 + j], gg = g[1024 + j], og = g[1536 + j];
    float c2 = sigmoidf(fg) * c[t] + sigmoidf(ig) * tanhf(gg);
    float h2 = sigmoidf(og) * tanhf(c2);
    c[t] = c2;
    h[t] = h2;
    if (ys) ys[t] = h2;
}

// ---------------------------------------------------------------------------
// Host side
// ---------------------------------------------------------------------------
static inline int nblk(long long total, int bs) { return (int)((total + bs - 1) / bs); }

#define GEMM_LAUNCH(FLAGS, A, lda, Bt, K, bias, C, ldc, M, N)                         \
    do {                                                                              \
        int groupsN_ = (N) / 64;                                                      \
        long long thr_ = (long long)((M) / 16) * groupsN_ * 32;                       \
        k_gemm_wmma<FLAGS><<<nblk(thr_, 256), 256, 0, stream>>>(                      \
            (A), (lda), (Bt), (K), (bias), (C), (ldc), (M), groupsN_);                \
    } while (0)

extern "C" void kernel_launch(void* const* d_in, const int* in_sizes, int n_in,
                              void* d_out, int out_size, void* d_ws, size_t ws_size,
                              hipStream_t stream) {
    (void)in_sizes; (void)n_in; (void)out_size; (void)ws_size;

    const int N = 20000, E = 320000, T = 16, B = 32;
    const int IN = 256, HID = 512, LAT = 128, G4 = 4 * HID;

    // ---- inputs (setup_inputs dict order) ----
    const float* x_seq = (const float*)d_in[0];
    const float* W1    = (const float*)d_in[1];
    const float* b1    = (const float*)d_in[2];
    const float* W2    = (const float*)d_in[3];
    const float* b2    = (const float*)d_in[4];
    const float* Wih0  = (const float*)d_in[5];
    const float* Whh0  = (const float*)d_in[6];
    const float* bih0  = (const float*)d_in[7];
    const float* bhh0  = (const float*)d_in[8];
    const float* Wih1  = (const float*)d_in[9];
    const float* Whh1  = (const float*)d_in[10];
    const float* bih1  = (const float*)d_in[11];
    const float* bhh1  = (const float*)d_in[12];
    const float* muW1  = (const float*)d_in[13];
    const float* mub1  = (const float*)d_in[14];
    const float* muW2  = (const float*)d_in[15];
    const float* mub2  = (const float*)d_in[16];
    const float* muW3  = (const float*)d_in[17];
    const float* mub3  = (const float*)d_in[18];
    const float* lvW1  = (const float*)d_in[19];
    const float* lvb1  = (const float*)d_in[20];
    const float* lvW2  = (const float*)d_in[21];
    const float* lvb2  = (const float*)d_in[22];
    const float* lvW3  = (const float*)d_in[23];
    const float* lvb3  = (const float*)d_in[24];
    const int*   edge  = (const int*)d_in[25];    // [2][E]
    const int*   batch = (const int*)d_in[26];    // [N]
    const int* src = edge;
    const int* dst = edge + E;

    float* out_mu = (float*)d_out;                       // [B][LAT]
    float* out_lv = out_mu + (size_t)B * LAT;
    float* out_z  = out_lv + (size_t)B * LAT;

    // ---- workspace carve-up ----
    size_t off = 0;
    auto carve = [&](size_t bytes) -> char* {
        char* p = (char*)d_ws + off;
        off += (bytes + 255) & ~(size_t)255;
        return p;
    };
    _Float16* W1t   = (_Float16*)carve((size_t)IN  * HID * 2);   // [HID][IN]
    _Float16* W2t   = (_Float16*)carve((size_t)HID * HID * 2);   // [HID][HID]
    _Float16* Wih0h = (_Float16*)carve((size_t)G4  * HID * 2);   // already [N][K]
    _Float16* Whh0h = (_Float16*)carve((size_t)G4  * HID * 2);
    _Float16* Wih1h = (_Float16*)carve((size_t)G4  * HID * 2);
    _Float16* Whh1h = (_Float16*)carve((size_t)G4  * HID * 2);
    _Float16* muW1t = (_Float16*)carve((size_t)HID * HID * 2);
    _Float16* muW2t = (_Float16*)carve((size_t)HID * HID * 2);
    _Float16* muW3t = (_Float16*)carve((size_t)HID * LAT * 2);
    _Float16* lvW1t = (_Float16*)carve((size_t)HID * HID * 2);
    _Float16* lvW2t = (_Float16*)carve((size_t)HID * HID * 2);
    _Float16* lvW3t = (_Float16*)carve((size_t)HID * LAT * 2);
    float* bsum0  = (float*)carve((size_t)G4 * 4);
    float* bsum1  = (float*)carve((size_t)G4 * 4);
    float* dinv   = (float*)carve((size_t)N * 4);
    float* nrm    = (float*)carve((size_t)E * 4);
    float* bufA   = (float*)carve((size_t)N * HID * 4);          // xw
    float* bufB   = (float*)carve((size_t)N * HID * 4);          // conv out / h
    float* pooled = (float*)carve((size_t)T * B * HID * 4);
    float* ys0    = (float*)carve((size_t)T * B * HID * 4);
    float* h0     = (float*)carve((size_t)B * HID * 4);
    float* c0     = (float*)carve((size_t)B * HID * 4);
    float* h1     = (float*)carve((size_t)B * HID * 4);
    float* c1     = (float*)carve((size_t)B * HID * 4);
    float* Gbuf   = (float*)carve((size_t)B * G4 * 4);
    float* tmp1   = (float*)carve((size_t)B * HID * 4);
    float* tmp2   = (float*)carve((size_t)B * HID * 4);

    // ---- weight prep (f32 -> f16, transposed to [N][K] where needed) ----
    k_cast_transpose_f16<<<nblk((long long)IN * HID, 256), 256, 0, stream>>>(W1, W1t, IN, HID);
    k_cast_transpose_f16<<<nblk((long long)HID * HID, 256), 256, 0, stream>>>(W2, W2t, HID, HID);
    k_cast_f16<<<nblk((long long)G4 * HID, 256), 256, 0, stream>>>(Wih0, Wih0h, (long long)G4 * HID);
    k_cast_f16<<<nblk((long long)G4 * HID, 256), 256, 0, stream>>>(Whh0, Whh0h, (long long)G4 * HID);
    k_cast_f16<<<nblk((long long)G4 * HID, 256), 256, 0, stream>>>(Wih1, Wih1h, (long long)G4 * HID);
    k_cast_f16<<<nblk((long long)G4 * HID, 256), 256, 0, stream>>>(Whh1, Whh1h, (long long)G4 * HID);
    k_cast_transpose_f16<<<nblk((long long)HID * HID, 256), 256, 0, stream>>>(muW1, muW1t, HID, HID);
    k_cast_transpose_f16<<<nblk((long long)HID * HID, 256), 256, 0, stream>>>(muW2, muW2t, HID, HID);
    k_cast_transpose_f16<<<nblk((long long)HID * LAT, 256), 256, 0, stream>>>(muW3, muW3t, HID, LAT);
    k_cast_transpose_f16<<<nblk((long long)HID * HID, 256), 256, 0, stream>>>(lvW1, lvW1t, HID, HID);
    k_cast_transpose_f16<<<nblk((long long)HID * HID, 256), 256, 0, stream>>>(lvW2, lvW2t, HID, HID);
    k_cast_transpose_f16<<<nblk((long long)HID * LAT, 256), 256, 0, stream>>>(lvW3, lvW3t, HID, LAT);
    k_vadd<<<nblk(G4, 256), 256, 0, stream>>>(bih0, bhh0, bsum0, G4);
    k_vadd<<<nblk(G4, 256), 256, 0, stream>>>(bih1, bhh1, bsum1, G4);

    // ---- degree / norm ----
    k_fill_f32<<<nblk(N, 256), 256, 0, stream>>>(dinv, 1.0f, N);      // deg = 1 (self loop)
    k_deg_edges<<<nblk(E, 256), 256, 0, stream>>>(dst, dinv, E);
    k_rsqrt_inplace<<<nblk(N, 256), 256, 0, stream>>>(dinv, N);
    k_edge_norm<<<nblk(E, 256), 256, 0, stream>>>(src, dst, dinv, nrm, E);

    // ---- GCN over time + pooling ----
    k_fill_f32<<<nblk((long long)T * B * HID, 256), 256, 0, stream>>>(pooled, 0.0f, (long long)T * B * HID);
    const long long NH  = (long long)N * HID;
    const long long EF  = (long long)E * 128;   // 4 features per thread
    for (int t = 0; t < T; ++t) {
        const float* xt = x_seq + (size_t)t * N * IN;
        GEMM_LAUNCH(0, xt, IN, W1t, IN, nullptr, bufA, HID, N, HID);
        k_conv_init<<<nblk(NH, 256), 256, 0, stream>>>(bufA, dinv, b1, bufB, NH);
        k_conv_edges<<<nblk(EF, 256), 256, 0, stream>>>(bufA, nrm, src, dst, bufB, EF);
        k_gelu_inplace<<<nblk(NH, 256), 256, 0, stream>>>(bufB, NH);
        GEMM_LAUNCH(0, bufB, HID, W2t, HID, nullptr, bufA, HID, N, HID);
        k_conv_init<<<nblk(NH, 256), 256, 0, stream>>>(bufA, dinv, b2, bufB, NH);
        k_conv_edges<<<nblk(EF, 256), 256, 0, stream>>>(bufA, nrm, src, dst, bufB, EF);
        k_pool<<<nblk(NH, 256), 256, 0, stream>>>(bufB, batch, pooled + (size_t)t * B * HID, NH);
    }

    // ---- LSTM layer 0 (keeps all timesteps) ----
    k_fill_f32<<<nblk(B * HID, 256), 256, 0, stream>>>(h0, 0.0f, B * HID);
    k_fill_f32<<<nblk(B * HID, 256), 256, 0, stream>>>(c0, 0.0f, B * HID);
    for (int t = 0; t < T; ++t) {
        const float* xt = pooled + (size_t)t * B * HID;
        GEMM_LAUNCH(GF_BIAS, xt, HID, Wih0h, HID, bsum0, Gbuf, G4, B, G4);
        GEMM_LAUNCH(GF_LOADC, h0, HID, Whh0h, HID, nullptr, Gbuf, G4, B, G4);
        k_lstm_gates<<<nblk(B * HID, 256), 256, 0, stream>>>(
            Gbuf, h0, c0, ys0 + (size_t)t * B * HID, B * HID);
    }
    // ---- LSTM layer 1 (only final h needed) ----
    k_fill_f32<<<nblk(B * HID, 256), 256, 0, stream>>>(h1, 0.0f, B * HID);
    k_fill_f32<<<nblk(B * HID, 256), 256, 0, stream>>>(c1, 0.0f, B * HID);
    for (int t = 0; t < T; ++t) {
        const float* xt = ys0 + (size_t)t * B * HID;
        GEMM_LAUNCH(GF_BIAS, xt, HID, Wih1h, HID, bsum1, Gbuf, G4, B, G4);
        GEMM_LAUNCH(GF_LOADC, h1, HID, Whh1h, HID, nullptr, Gbuf, G4, B, G4);
        k_lstm_gates<<<nblk(B * HID, 256), 256, 0, stream>>>(Gbuf, h1, c1, nullptr, B * HID);
    }

    // ---- MLP heads ----
    GEMM_LAUNCH(GF_BIAS | GF_GELU, h1,   HID, muW1t, HID, mub1, tmp1, HID, B, HID);
    GEMM_LAUNCH(GF_BIAS | GF_GELU, tmp1, HID, muW2t, HID, mub2, tmp2, HID, B, HID);
    GEMM_LAUNCH(GF_BIAS,           tmp2, HID, muW3t, HID, mub3, out_mu, LAT, B, LAT);
    GEMM_LAUNCH(GF_BIAS | GF_GELU, h1,   HID, lvW1t, HID, lvb1, tmp1, HID, B, HID);
    GEMM_LAUNCH(GF_BIAS | GF_GELU, tmp1, HID, lvW2t, HID, lvb2, tmp2, HID, B, HID);
    GEMM_LAUNCH(GF_BIAS | GF_CLAMP10, tmp2, HID, lvW3t, HID, lvb3, out_lv, LAT, B, LAT);
    k_copy_f32<<<nblk(B * LAT, 256), 256, 0, stream>>>(out_mu, out_z, B * LAT);
}
